// MambaBlock_59450937312021
// MI455X (gfx1250) — matrix operand order repaired and checked by reading, not compiled, not run on hardware
//
#include <hip/hip_runtime.h>
#include <hip/hip_bf16.h>
#include <cstddef>

// ---------------- problem constants ----------------
#define D_MODEL  1024
#define D_STATE  16
#define D_CONV   4
#define D_INNER  2048           // EXPAND * D_MODEL
#define DT_RANK  64             // ceil(1024/16)
#define BATCH    2
#define LENGTH   2048
#define MROWS    (BATCH * LENGTH)   // 4096
#define XDBL_W   (DT_RANK + 2 * D_STATE)  // 96

// ---------------- WMMA types ----------------
typedef __bf16 bf16_t;
typedef __attribute__((ext_vector_type(16))) __bf16 v16bf;
typedef __attribute__((ext_vector_type(8)))  __bf16 v8bf;
typedef __attribute__((ext_vector_type(4)))  __bf16 v4bf;
typedef __attribute__((ext_vector_type(8)))  float  v8f;

// ---------------- generic bf16 WMMA GEMM: C = A(fp32) @ B(fp32) ----------------
// Assumes M % 128 == 0 and K % 32 == 0 (true for all launches here).
// BFULL: N edge handling compiled only when needed (x_proj, N=96).
// Block: 256 threads (8 waves). Tile: 128(M) x 128(N), K step 32.
// Wave (wm in 0..1, wn in 0..3) computes a 64x32 subtile = 4x2 WMMA accumulators.
// LDS layouts give each lane a contiguous, 16B-aligned fragment (2x ds_load_b128):
//   As[m][40]   row-major,  A-frag lane reads k = {akb..akb+7, akb+16..akb+23}
//   Bs_t[n][40] transposed, B-frag lane reads k = {bkb..bkb+7, bkb+16..bkb+23}
#define TM 128
#define TN 128
#define TK 32
#define LDK 40   // padded K stride: 80 bytes, multiple of 16

template <bool BFULL>
__global__ __launch_bounds__(256) void gemm_bf16_wmma(
    const float* __restrict__ A, int lda,
    const float* __restrict__ B, int ldb,
    float* __restrict__ C, int ldc,
    int M, int N, int K)
{
    __shared__ __align__(16) bf16_t As[TM][LDK];     // 10 KB
    __shared__ __align__(16) bf16_t Bs_t[TN][LDK];   // 10 KB

    const int t    = threadIdx.x;
    const int lane = t & 31;
    const int wave = t >> 5;
    const int wm   = wave >> 2;          // 0..1 -> 64 rows each
    const int wn   = wave & 3;           // 0..3 -> 32 cols each

    const int tileM = blockIdx.y * TM;
    const int tileN = blockIdx.x * TN;

    v8f acc[4][2];
#pragma unroll
    for (int mt = 0; mt < 4; ++mt)
#pragma unroll
        for (int nt = 0; nt < 2; ++nt)
            acc[mt][nt] = (v8f){0.f, 0.f, 0.f, 0.f, 0.f, 0.f, 0.f, 0.f};

    for (int k0 = 0; k0 < K; k0 += TK) {
        // ---- stage A tile (128x32): batch 4x global_load_b128, then cvt+store ----
        {
            float4 fa[4];
#pragma unroll
            for (int i = 0; i < 4; ++i) {            // 1024 float4 / 256 thr
                int vi = i * 256 + t;
                int r  = vi >> 3;                    // 8 float4 per 32-wide row
                int c4 = vi & 7;
                unsigned aoff = (unsigned)(tileM + r) * (unsigned)lda
                              + (unsigned)(k0 + c4 * 4);
                fa[i] = *(const float4*)(A + aoff);
            }
#pragma unroll
            for (int i = 0; i < 4; ++i) {
                int vi = i * 256 + t;
                int r  = vi >> 3;
                int c4 = vi & 7;
                v4bf p = { (bf16_t)fa[i].x, (bf16_t)fa[i].y,
                           (bf16_t)fa[i].z, (bf16_t)fa[i].w };
                *(v4bf*)(&As[r][c4 * 4]) = p;        // 8B aligned
            }
        }

        // ---- stage B tile (32x128) as bf16, transposed Bs_t[n][k] ----
        if (BFULL) {
            int n  = t & 127;
            int kh = (t >> 7) * 16;                  // 0 or 16
            unsigned boff = (unsigned)(k0 + kh) * (unsigned)ldb + (unsigned)(tileN + n);
            float fb[16];
#pragma unroll
            for (int kk = 0; kk < 16; ++kk)
                fb[kk] = B[boff + (unsigned)(kk * ldb)];
#pragma unroll
            for (int kk = 0; kk < 16; kk += 4) {
                v4bf p = { (bf16_t)fb[kk], (bf16_t)fb[kk + 1],
                           (bf16_t)fb[kk + 2], (bf16_t)fb[kk + 3] };
                *(v4bf*)(&Bs_t[n][kh + kk]) = p;     // 8B aligned
            }
        } else {
#pragma unroll
            for (int i = 0; i < (TK * TN) / 256; ++i) {
                int idx = i * 256 + t;
                int r = idx >> 7, c = idx & (TN - 1);
                int gc = tileN + c;
                float v = (gc < N) ? B[(unsigned)(k0 + r) * (unsigned)ldb + (unsigned)gc]
                                   : 0.f;
                Bs_t[c][r] = (bf16_t)v;
            }
        }
        __syncthreads();

        // ---- prefetch next K tile into cache while we compute ----
        if (BFULL && (k0 + TK < K)) {
            int pr = t >> 1;
            __builtin_prefetch(A + (unsigned)(tileM + pr) * (unsigned)lda
                                 + (unsigned)(k0 + TK + (t & 1) * 16), 0, 0);
            if (t < TK)
                __builtin_prefetch(B + (unsigned)(k0 + TK + t) * (unsigned)ldb
                                     + (unsigned)tileN, 0, 0);
        }

        // ---- build fragments: 2x ds_load_b128 each, no shuffles ----
        v16bf afrag[4], bfrag[2];
        const int mrow = lane & 15;
        const int akb  = (lane >> 4) << 3;    // 0 or 8
#pragma unroll
        for (int mt = 0; mt < 4; ++mt) {
            int m = wm * 64 + mt * 16 + mrow;
            const bf16_t* ap = &As[m][akb];
            v8bf lo = *(const v8bf*)(ap);         // k = akb .. akb+7
            v8bf hi = *(const v8bf*)(ap + 16);    // k = akb+16 .. akb+23
            afrag[mt] = __builtin_shufflevector(lo, hi,
                0, 1, 2, 3, 4, 5, 6, 7, 8, 9, 10, 11, 12, 13, 14, 15);
        }
        const int ncol = lane & 15;
        const int bkb  = (lane >> 4) << 3;    // 0 or 8
#pragma unroll
        for (int nt = 0; nt < 2; ++nt) {
            int n = wn * 32 + nt * 16 + ncol;
            const bf16_t* bp = &Bs_t[n][bkb];
            v8bf lo = *(const v8bf*)(bp);         // k = bkb .. bkb+7
            v8bf hi = *(const v8bf*)(bp + 16);    // k = bkb+16 .. bkb+23
            bfrag[nt] = __builtin_shufflevector(lo, hi,
                0, 1, 2, 3, 4, 5, 6, 7, 8, 9, 10, 11, 12, 13, 14, 15);
        }

        // ---- 8 WMMAs per K-step ----
#pragma unroll
        for (int mt = 0; mt < 4; ++mt)
#pragma unroll
            for (int nt = 0; nt < 2; ++nt)
                acc[mt][nt] = __builtin_amdgcn_wmma_f32_16x16x32_bf16(
                    false, afrag[mt], false, bfrag[nt],
                    (short)0, acc[mt][nt], false, false);

        __syncthreads();
    }

    // ---- store C (fp32). C/D layout: element r -> row r + 8*(lane>=16), col lane&15 ----
    const int rbase = (lane >> 4) << 3;
    const int ccol  = lane & 15;
#pragma unroll
    for (int mt = 0; mt < 4; ++mt) {
#pragma unroll
        for (int nt = 0; nt < 2; ++nt) {
            int col = tileN + wn * 32 + nt * 16 + ccol;
            if (BFULL || col < N) {
#pragma unroll
                for (int r = 0; r < 8; ++r) {
                    int row = tileM + wm * 64 + mt * 16 + rbase + r;
                    C[(unsigned)row * (unsigned)ldc + (unsigned)col] = acc[mt][nt][r];
                }
            }
        }
    }
}

// ---------------- depthwise causal conv (k=4) + bias + SiLU ----------------
__global__ __launch_bounds__(256) void conv_silu_kernel(
    const float* __restrict__ xr, const float* __restrict__ w,
    const float* __restrict__ bias, float* __restrict__ xc)
{
    unsigned idx = blockIdx.x * blockDim.x + threadIdx.x;
    if (idx >= MROWS * D_INNER) return;
    unsigned d  = idx & (D_INNER - 1);
    unsigned bl = idx / D_INNER;
    int      l  = (int)(bl & (LENGTH - 1));
    unsigned b  = bl / LENGTH;
    float acc = bias[d];
#pragma unroll
    for (int j = 0; j < D_CONV; ++j) {
        int lj = l + j - (D_CONV - 1);
        if (lj >= 0)
            acc += xr[(b * LENGTH + (unsigned)lj) * (2 * D_INNER) + d] * w[d * D_CONV + j];
    }
    float s = 1.f / (1.f + __expf(-acc));
    xc[bl * D_INNER + d] = acc * s;
}

// ---------------- dt = softplus(dt_lin + dt_proj_b) in place ----------------
__global__ __launch_bounds__(256) void softplus_bias_kernel(
    float* __restrict__ dt, const float* __restrict__ b)
{
    unsigned idx = blockIdx.x * blockDim.x + threadIdx.x;
    if (idx >= MROWS * D_INNER) return;
    unsigned d = idx & (D_INNER - 1);
    float v = dt[idx] + b[d];
    dt[idx] = (v > 20.f) ? v : __logf(1.f + __expf(v));
}

// ---------------- selective scan + D skip + gating with silu(res) ----------------
// One thread per (b, d) channel; 16 states + 16 A values in registers.
// y overwrites xc in place (each element read once by its owner thread first).
__global__ __launch_bounds__(256) void scan_kernel(
    const float* xc_in, const float* __restrict__ dt,
    const float* __restrict__ xdbl, const float* __restrict__ A_log,
    const float* __restrict__ Dp, const float* __restrict__ xr,
    float* y_out)
{
    unsigned gid = blockIdx.x * blockDim.x + threadIdx.x;
    if (gid >= BATCH * D_INNER) return;
    unsigned d = gid & (D_INNER - 1);
    unsigned b = gid / D_INNER;

    float A[D_STATE], s[D_STATE];
#pragma unroll
    for (int n = 0; n < D_STATE; ++n) {
        A[n] = -__expf(A_log[d * D_STATE + n]);
        s[n] = 0.f;
    }
    const float Dd = Dp[d];

    for (int l = 0; l < LENGTH; ++l) {
        unsigned row = b * LENGTH + (unsigned)l;
        float xv  = xc_in[row * D_INNER + d];
        float dtv = dt[row * D_INNER + d];
        const float* bc = xdbl + row * XDBL_W;
        float yv = 0.f;
        float dx = dtv * xv;
#pragma unroll
        for (int n = 0; n < D_STATE; ++n) {
            s[n] = s[n] * __expf(dtv * A[n]) + dx * bc[DT_RANK + n];
            yv  += s[n] * bc[DT_RANK + D_STATE + n];
        }
        yv += xv * Dd;
        float r = xr[row * (2 * D_INNER) + D_INNER + d];   // res
        yv *= r * (1.f / (1.f + __expf(-r)));              // y * silu(res)
        y_out[row * D_INNER + d] = yv;
    }
}

// ---------------- launcher ----------------
extern "C" void kernel_launch(void* const* d_in, const int* in_sizes, int n_in,
                              void* d_out, int out_size, void* d_ws, size_t ws_size,
                              hipStream_t stream) {
    const float* x          = (const float*)d_in[0];
    const float* in_proj_w  = (const float*)d_in[1];
    const float* conv_w     = (const float*)d_in[2];
    const float* conv_b     = (const float*)d_in[3];
    const float* x_proj_w   = (const float*)d_in[4];
    const float* dt_proj_w  = (const float*)d_in[5];
    const float* dt_proj_b  = (const float*)d_in[6];
    const float* A_log      = (const float*)d_in[7];
    const float* Dparam     = (const float*)d_in[8];
    const float* out_proj_w = (const float*)d_in[9];
    float* out = (float*)d_out;

    char* ws = (char*)d_ws;
    float* xr   = (float*)(ws);                               // 4096 x 4096  (64 MB)
    float* xc   = (float*)(ws + ((size_t)64 << 20));          // 4096 x 2048  (32 MB)
    float* xdbl = (float*)(ws + ((size_t)96 << 20));          // 4096 x 96    (1.5 MB)
    float* dtb  = (float*)(ws + ((size_t)98 << 20));          // 4096 x 2048  (32 MB)

    const int elems = MROWS * D_INNER;
    dim3 blk(256);

    // 1. xr = x @ in_proj_w            (4096 x 4096 x 1024)
    gemm_bf16_wmma<true><<<dim3((2 * D_INNER) / TN, MROWS / TM), blk, 0, stream>>>(
        x, D_MODEL, in_proj_w, 2 * D_INNER, xr, 2 * D_INNER,
        MROWS, 2 * D_INNER, D_MODEL);

    // 2. xc = silu(depthwise_conv(xr[:, :D_INNER]) + conv_b)
    conv_silu_kernel<<<(elems + 255) / 256, blk, 0, stream>>>(xr, conv_w, conv_b, xc);

    // 3. xdbl = xc @ x_proj_w          (4096 x 96 x 2048), N edge -> BFULL=false
    gemm_bf16_wmma<false><<<dim3((XDBL_W + TN - 1) / TN, MROWS / TM), blk, 0, stream>>>(
        xc, D_INNER, x_proj_w, XDBL_W, xdbl, XDBL_W,
        MROWS, XDBL_W, D_INNER);

    // 4. dt_lin = xdbl[:, :DT_RANK] @ dt_proj_w   (4096 x 2048 x 64)
    gemm_bf16_wmma<true><<<dim3(D_INNER / TN, MROWS / TM), blk, 0, stream>>>(
        xdbl, XDBL_W, dt_proj_w, D_INNER, dtb, D_INNER,
        MROWS, D_INNER, DT_RANK);

    // 5. dt = softplus(dt_lin + dt_proj_b)  (in place)
    softplus_bias_kernel<<<(elems + 255) / 256, blk, 0, stream>>>(dtb, dt_proj_b);

    // 6. selective scan + D skip + silu(res) gating; y overwrites xc in place
    scan_kernel<<<(BATCH * D_INNER) / 256, blk, 0, stream>>>(
        xc, dtb, xdbl, A_log, Dparam, xr, xc);

    // 7. out = y @ out_proj_w          (4096 x 1024 x 2048)
    gemm_bf16_wmma<true><<<dim3(D_MODEL / TN, MROWS / TM), blk, 0, stream>>>(
        xc, D_INNER, out_proj_w, D_MODEL, out, D_MODEL,
        MROWS, D_MODEL, D_INNER);
}
